// RWKV_Block_2078764171748
// MI455X (gfx1250) — matrix operand order repaired and checked by reading, not compiled, not run on hardware
//
#include <hip/hip_runtime.h>
#include <math.h>

// Problem constants (fixed by the reference harness)
#define N_EMBD 2048
#define SEQ 2048
#define BATCH 4
#define NROWS (BATCH * SEQ)       // 8192 token rows
#define STATE_ROWS 66             // 2 + HEAD_SIZE
#define W1_COLS 160
#define OUT0 ((size_t)NROWS * 5 * N_EMBD)  // elements of x_kwvrg before new_state

typedef __attribute__((ext_vector_type(16))) __bf16          bf16x16;
typedef __attribute__((ext_vector_type(8)))  float           f32x8;
typedef __attribute__((ext_vector_type(4)))  float           f32x4;
typedef __attribute__((ext_vector_type(8)))  unsigned short  u16x8;
typedef __attribute__((ext_vector_type(4)))  unsigned short  u16x4;

union BF16Frag { bf16x16 v; u16x8 v8[2]; unsigned short u[16]; };
union AccFrag  { f32x8 v;   float f[8]; };

__device__ __forceinline__ unsigned short f2bf(float f) {
    unsigned int u = __builtin_bit_cast(unsigned int, f);
    unsigned int r = u + 0x7FFFu + ((u >> 16) & 1u);  // round-to-nearest-even
    return (unsigned short)(r >> 16);
}

// row -> pointer to the "previous token" row (state row for t==0)
__device__ __forceinline__ const float* prev_row_ptr(const float* x,
                                                     const float* state,
                                                     int R, int i1) {
    int b = R >> 11, t = R & (SEQ - 1);
    return (t == 0) ? state + ((size_t)(b * STATE_ROWS + i1)) * N_EMBD
                    : x + (size_t)(R - 1) * N_EMBD;
}

// ---------------------------------------------------------------------------
// Kernel 0: bf16-transpose W1 -> w1t[160][2048], W2 -> w2t[5][2048][32];
//           produce new_state output.
// ---------------------------------------------------------------------------
__global__ void rwkv_prep_kernel(const float* __restrict__ x,
                                 const float* __restrict__ state,
                                 const float* __restrict__ w1,
                                 const float* __restrict__ w2,
                                 const int* __restrict__ ip,
                                 unsigned short* __restrict__ w1t,
                                 unsigned short* __restrict__ w2t,
                                 float* __restrict__ out_state) {
    const int i1 = STATE_ROWS * ip[0] + 1;
    const int stride = gridDim.x * blockDim.x;
    const int tid = blockIdx.x * blockDim.x + threadIdx.x;
    // w1t[col][k] = bf16(w1[k][col])
    for (int idx = tid; idx < W1_COLS * N_EMBD; idx += stride) {
        int col = idx >> 11, k = idx & (N_EMBD - 1);
        w1t[idx] = f2bf(w1[k * W1_COLS + col]);
    }
    // w2t[f][c][k] = bf16(w2[f][k][c])
    for (int idx = tid; idx < 5 * N_EMBD * 32; idx += stride) {
        int f = idx >> 16;               // 2048*32 = 65536
        int rem = idx & 65535;
        int c = rem >> 5, k = rem & 31;
        w2t[idx] = f2bf(w2[((size_t)(f * 32 + k)) * N_EMBD + c]);
    }
    // new_state = state with row i1 <- x[:, S-1]
    for (int idx = tid; idx < BATCH * STATE_ROWS * N_EMBD; idx += stride) {
        int d   = idx % N_EMBD;
        int row = (idx / N_EMBD) % STATE_ROWS;
        int b   = idx / (N_EMBD * STATE_ROWS);
        float v = (row == i1) ? x[((size_t)(b * SEQ + (SEQ - 1))) * N_EMBD + d]
                              : state[idx];
        out_state[idx] = v;
    }
}

// ---------------------------------------------------------------------------
// Kernel 1: xxx = tanh( (x + sx*tmx) @ W1 ), stored bf16 [NROWS x 160].
// Block: 256 thr (8 waves), tile M=64 x N=160, K step 32.
// LDS: A row-major (K contig, 80B row stride), B transposed (K contig per col)
// so every fragment is 2x ds_load_b128.
// ---------------------------------------------------------------------------
__global__ __launch_bounds__(256) void rwkv_gemm1_kernel(
    const float* __restrict__ x, const float* __restrict__ state,
    const float* __restrict__ tmx, const unsigned short* __restrict__ w1t,
    const int* __restrict__ ip, unsigned short* __restrict__ xxx) {
    __shared__ unsigned short lds_a[64][40];     // xin tile, bf16, 80B stride
    __shared__ unsigned short lds_bT[W1_COLS][40]; // W1 tile transposed

    const int i1   = STATE_ROWS * ip[0] + 1;
    const int tid  = threadIdx.x;
    const int lane = tid & 31;
    const int w    = tid >> 5;
    const int wm   = w >> 1;         // 0..3 -> 16-row slice
    const int wn   = w & 1;          // 0..1 -> 80-col slice
    const int r0   = blockIdx.x * 64;
    const int half = lane >> 4;
    const int nl   = lane & 15;

    // staging roles: each thread fills rows {arow, arow+32}, cols [acol4, +4)
    const int arow  = tid >> 3;      // 0..31
    const int acol4 = (tid & 7) * 4;

    const float* pxa = x + (size_t)(r0 + arow) * N_EMBD + acol4;
    const float* ppa = prev_row_ptr(x, state, r0 + arow, i1) + acol4;
    const float* pxb = x + (size_t)(r0 + arow + 32) * N_EMBD + acol4;
    const float* ppb = prev_row_ptr(x, state, r0 + arow + 32, i1) + acol4;
    const float* ptm = tmx + acol4;

    const f32x8 zacc = {0.f, 0.f, 0.f, 0.f, 0.f, 0.f, 0.f, 0.f};
    AccFrag acc[5];
#pragma unroll
    for (int n = 0; n < 5; ++n) acc[n].v = zacc;

    for (int k0 = 0; k0 < N_EMBD; k0 += 32) {
        // Stage A: xin = x + (xprev - x)*tmx  (f32x4 loads, u16x4 LDS store)
        f32x4 tm = *(const f32x4*)(ptm + k0);
#pragma unroll
        for (int rr = 0; rr < 2; ++rr) {
            f32x4 xv = *(const f32x4*)((rr ? pxb : pxa) + k0);
            f32x4 xp = *(const f32x4*)((rr ? ppb : ppa) + k0);
            u16x4 pk;
#pragma unroll
            for (int j = 0; j < 4; ++j)
                pk[j] = f2bf(xv[j] + (xp[j] - xv[j]) * tm[j]);
            *(u16x4*)&lds_a[arow + rr * 32][acol4] = pk;
        }
        // Stage B: 160 cols x 32 k as 640 chunks of 8 ushorts (b128 both ends)
#pragma unroll
        for (int c = tid; c < 640; c += 256) {
            int col = c >> 2, part = c & 3;
            *(u16x8*)&lds_bT[col][part * 8] =
                *(const u16x8*)(w1t + (size_t)col * N_EMBD + k0 + part * 8);
        }
        __syncthreads();

        // A fragment: two contiguous 16B runs
        BF16Frag a;
        a.v8[0] = *(const u16x8*)&lds_a[wm * 16 + nl][half * 8];
        a.v8[1] = *(const u16x8*)&lds_a[wm * 16 + nl][16 + half * 8];
#pragma unroll
        for (int nt = 0; nt < 5; ++nt) {
            const int col = wn * 80 + nt * 16 + nl;
            BF16Frag b;   // lanes 0-15: K 0..7,16..23; lanes 16-31: K 8..15,24..31
            b.v8[0] = *(const u16x8*)&lds_bT[col][half * 8];
            b.v8[1] = *(const u16x8*)&lds_bT[col][16 + half * 8];
            acc[nt].v = __builtin_amdgcn_wmma_f32_16x16x32_bf16(
                false, a.v, false, b.v, (short)0, acc[nt].v, false, false);
        }
        __syncthreads();
    }

    // Epilogue: tanh, pack bf16. D layout: VGPR v -> M = v + 8*half
#pragma unroll
    for (int nt = 0; nt < 5; ++nt) {
        const int col = wn * 80 + nt * 16 + nl;
#pragma unroll
        for (int v = 0; v < 8; ++v) {
            int row = r0 + wm * 16 + half * 8 + v;
            xxx[(size_t)row * W1_COLS + col] = f2bf(tanhf(acc[nt].f[v]));
        }
    }
}

// ---------------------------------------------------------------------------
// Kernel 2: out[b,t,f,:] = x + sx*(maa_f + xxx[:,f*32:(f+1)*32] @ W2[f])
// Block: 256 thr (8 waves), tile M=32 rows x N=64 cols, all 5 modes.
// ---------------------------------------------------------------------------
__global__ __launch_bounds__(256) void rwkv_gemm2_kernel(
    const float* __restrict__ x, const float* __restrict__ state,
    const unsigned short* __restrict__ w2t,   // [5][2048][32] bf16, K contig
    const unsigned short* __restrict__ xxx,   // [NROWS][160] bf16
    const float* __restrict__ maa_k, const float* __restrict__ maa_w,
    const float* __restrict__ maa_v, const float* __restrict__ maa_r,
    const float* __restrict__ maa_g,
    const int* __restrict__ ip, float* __restrict__ out) {
    __shared__ unsigned short lds_a[32][168];     // xxx tile (K contig, 336B)
    __shared__ unsigned short lds_bT[5][64][40];  // W2 tiles, K contig per col
    __shared__ float lds_x[32][64];
    __shared__ float lds_sx[32][64];

    const int i1   = STATE_ROWS * ip[0] + 1;
    const int tid  = threadIdx.x;
    const int lane = tid & 31;
    const int w    = tid >> 5;
    const int wm   = w >> 2;        // 0..1
    const int wn   = w & 3;         // 0..3
    const int r0   = blockIdx.y * 32;
    const int c0   = blockIdx.x * 64;
    const int half = lane >> 4;
    const int nl   = lane & 15;

    // Stage xxx tile: 32 rows x 160 = 640 chunks of 8 ushorts
#pragma unroll
    for (int c = tid; c < 640; c += 256) {
        int row = c / 20, part = c % 20;
        *(u16x8*)&lds_a[row][part * 8] =
            *(const u16x8*)(xxx + (size_t)(r0 + row) * W1_COLS + part * 8);
    }
    // Stage W2: 5 f x 64 cols x 32 k = 1280 chunks of 8 ushorts
#pragma unroll
    for (int c = tid; c < 1280; c += 256) {
        int f = c >> 8;                  // 64*4 = 256 chunks per f
        int rem = c & 255;
        int col = rem >> 2, part = rem & 3;
        *(u16x8*)&lds_bT[f][col][part * 8] =
            *(const u16x8*)(w2t + (((size_t)f * N_EMBD) + c0 + col) * 32 + part * 8);
    }
    // Stage x / sx fp32 tiles: 512 f32x4 slots
#pragma unroll
    for (int s = tid; s < 512; s += 256) {
        int row = s >> 4, cg = (s & 15) * 4;
        int R = r0 + row;
        f32x4 xv = *(const f32x4*)(x + (size_t)R * N_EMBD + c0 + cg);
        f32x4 xp = *(const f32x4*)(prev_row_ptr(x, state, R, i1) + c0 + cg);
#pragma unroll
        for (int j = 0; j < 4; ++j) {
            lds_x[row][cg + j]  = xv[j];
            lds_sx[row][cg + j] = xp[j] - xv[j];
        }
    }
    __syncthreads();

    const int colL = wn * 16 + nl;  // local column 0..63
    const int colG = c0 + colL;
    const f32x8 zacc = {0.f, 0.f, 0.f, 0.f, 0.f, 0.f, 0.f, 0.f};

#pragma unroll
    for (int f = 0; f < 5; ++f) {
        BF16Frag a, b;
        a.v8[0] = *(const u16x8*)&lds_a[wm * 16 + nl][f * 32 + half * 8];
        a.v8[1] = *(const u16x8*)&lds_a[wm * 16 + nl][f * 32 + 16 + half * 8];
        b.v8[0] = *(const u16x8*)&lds_bT[f][colL][half * 8];
        b.v8[1] = *(const u16x8*)&lds_bT[f][colL][16 + half * 8];

        AccFrag acc;
        acc.v = __builtin_amdgcn_wmma_f32_16x16x32_bf16(
            false, a.v, false, b.v, (short)0, zacc, false, false);

        const float* maap = (f == 0) ? maa_k : (f == 1) ? maa_w
                          : (f == 2) ? maa_v : (f == 3) ? maa_r : maa_g;
        const float st = maap[colG];
#pragma unroll
        for (int v = 0; v < 8; ++v) {
            int rowL = wm * 16 + half * 8 + v;
            int R = r0 + rowL;
            float xv  = lds_x[rowL][colL];
            float sxv = lds_sx[rowL][colL];
            out[((size_t)R * 5 + f) * N_EMBD + colG] = xv + sxv * (st + acc.f[v]);
        }
    }
}

extern "C" void kernel_launch(void* const* d_in, const int* in_sizes, int n_in,
                              void* d_out, int out_size, void* d_ws, size_t ws_size,
                              hipStream_t stream) {
    (void)in_sizes; (void)n_in; (void)out_size; (void)ws_size;
    const float* x     = (const float*)d_in[0];
    const float* state = (const float*)d_in[1];
    const float* tmx   = (const float*)d_in[2];
    const float* w1    = (const float*)d_in[3];
    const float* w2    = (const float*)d_in[4];
    const float* maa_k = (const float*)d_in[5];
    const float* maa_w = (const float*)d_in[6];
    const float* maa_v = (const float*)d_in[7];
    const float* maa_r = (const float*)d_in[8];
    const float* maa_g = (const float*)d_in[9];
    const int*   ip    = (const int*)d_in[10];

    float* out       = (float*)d_out;
    float* out_state = out + OUT0;

    unsigned short* w1t = (unsigned short*)d_ws;                  // 160*2048
    unsigned short* w2t = w1t + (size_t)W1_COLS * N_EMBD;         // 5*2048*32
    unsigned short* xxx = w2t + (size_t)5 * N_EMBD * 32;          // 8192*160

    rwkv_prep_kernel<<<512, 256, 0, stream>>>(x, state, w1, w2, ip,
                                              w1t, w2t, out_state);

    rwkv_gemm1_kernel<<<NROWS / 64, 256, 0, stream>>>(x, state, tmx, w1t,
                                                      ip, xxx);

    dim3 g2(N_EMBD / 64, NROWS / 32);
    rwkv_gemm2_kernel<<<g2, 256, 0, stream>>>(x, state, w2t, xxx,
                                              maa_k, maa_w, maa_v, maa_r, maa_g,
                                              ip, out);
}
